// MoeLayer_31430570672220
// MI455X (gfx1250) — compile-verified, matrix operand order checked by reading, not run on hardware
//
#include <hip/hip_runtime.h>
#include <hip/hip_bf16.h>
#include <math.h>

// ---------------------------------------------------------------------------
// MoE dense layer for MI455X (gfx1250, wave32, WMMA f32_16x16x32_f16 + TDM)
//   out[n,o] = sum_e gate[n,e] * ( x[n,:] @ W[e,:,o] + b[e,o] )
// Gate-folded single-accumulator GEMM. A fragments (x) are expert-invariant:
// hoisted to registers per K-chunk, scaled per expert by one f16 scalar/lane.
// All global->LDS tile movement is done by the Tensor Data Mover
// (tensor_load_to_lds, TENSORcnt), with the TDM pad feature reproducing the
// padded LDS pitch (64 halfs + 8 pad = 144B rows). One barrier per iteration.
// Workspace layout (~33.8 MB):
//   [0,   16MB)  W transposed+converted: Wt[e][o][k] f16
//   [16MB,32MB)  x converted: xh[n][k] f16
//   [32MB,+256K) gate softmax scores f32 [n][8]
// ---------------------------------------------------------------------------

typedef __attribute__((ext_vector_type(16))) _Float16 v16h;
typedef __attribute__((ext_vector_type(8)))  _Float16 v8h;
typedef __attribute__((ext_vector_type(4)))  _Float16 v4h;
typedef __attribute__((ext_vector_type(8)))  float    v8f;
typedef __attribute__((ext_vector_type(4)))  unsigned v4u;
typedef __attribute__((ext_vector_type(8)))  int      v8i;
typedef __attribute__((ext_vector_type(4)))  int      v4i;

#define N_TOKENS 8192
#define D_IN     1024
#define D_OUT    1024
#define N_EXP    8

#define BM 128
#define BN 128
#define BK 64
#define PK 72   // LDS K-pitch in halfs: 128B data + 16B pad = 144B rows

static __device__ __forceinline__ v16h join16(v8h lo, v8h hi) {
    v16h r;
#pragma unroll
    for (int i = 0; i < 8; ++i) { r[i] = lo[i]; r[i + 8] = hi[i]; }
    return r;
}

// ---- TDM: DMA a 128-row x 64-half tile (row stride 1024 halfs) into LDS,
// padding +16B per 128B row so rows land on the PK=72-half pitch.
// D# layout per CDNA5 ISA.
// This toolchain's builtin takes 6 args: (g0, g1, g2, g3, g4, cpol).
static __device__ __forceinline__ void tdm_load_tile_128x64(const _Float16* src,
                                                            unsigned lds_addr) {
    const unsigned long long ga = (unsigned long long)(size_t)src;
    v4u g0;
    g0[0] = 1u;                                      // count=1 (valid user D#)
    g0[1] = lds_addr;                                // LDS byte address
    g0[2] = (unsigned)ga;                            // global_addr[31:0]
    g0[3] = (unsigned)((ga >> 32) & 0x01FFFFFFu)     // global_addr[56:32]
          | (2u << 30);                              // type=2 ("image")
    v8i g1;
    g1[0] = (int)((1u << 16)      // data_size=1 -> 2 bytes
                | (1u << 20)      // pad_enable
                | (4u << 22)      // pad_interval=4 -> every 32 DWORDs (128B)
                | (3u << 25));    // pad_amount=3  -> 4 DWORDs (16B)
    g1[1] = (int)(1024u << 16);   // tensor_dim0 = 1024 (low16 -> bits[31:16])
    g1[2] = 0;                    // tensor_dim0 hi16 = 0, tensor_dim1 lo16 = 0
    g1[3] = (int)(16u | (64u << 16));  // tensor_dim1 = 1<<20; tile_dim0 = 64
    g1[4] = 128;                  // tile_dim1 = 128, tile_dim2 = 0
    g1[5] = 1024;                 // tensor_dim0_stride[31:0] = 1024 elements
    g1[6] = 0;                    // stride0 hi, tensor_dim1_stride lo16
    g1[7] = 0;
    const v4i g2 = {0, 0, 0, 0};  // 2-D tensor: higher dims unused
    const v4i g3 = {0, 0, 0, 0};
    const v8i g4 = {0, 0, 0, 0, 0, 0, 0, 0};
    __builtin_amdgcn_tensor_load_to_lds(g0, g1, g2, g3, g4, 0);
}

// --------------------------- f32 -> f16 convert (row-major copy) -----------
__global__ __launch_bounds__(256)
void cvt_f32_to_f16(const float* __restrict__ src, _Float16* __restrict__ dst, int n4) {
    int i = blockIdx.x * 256 + threadIdx.x;
    if (i < n4) {
        float4 v = ((const float4*)src)[i];
        v4h o;
        o[0] = (_Float16)v.x; o[1] = (_Float16)v.y;
        o[2] = (_Float16)v.z; o[3] = (_Float16)v.w;
        ((v4h*)dst)[i] = o;
    }
}

// ------------------ W: f32 [e][k][n] -> f16 transposed [e][n][k] -----------
__global__ __launch_bounds__(256)
void transpose_cvt_w(const float* __restrict__ W, _Float16* __restrict__ Wt) {
    const int e  = blockIdx.z;
    const int tn = blockIdx.x;   // 64-wide n tile
    const int tk = blockIdx.y;   // 64-wide k tile
    __shared__ _Float16 t[64][72];
    const int tid = threadIdx.x;
    const float* Wp = W + ((size_t)e << 20);
#pragma unroll
    for (int i = 0; i < 16; ++i) {
        int lin = tid + i * 256;              // 0..4095: r=k idx, c=n idx
        int r = lin >> 6, c = lin & 63;
        t[c][r] = (_Float16)Wp[(size_t)(tk * 64 + r) * D_OUT + tn * 64 + c];
    }
    __syncthreads();
    _Float16* Wtp = Wt + ((size_t)e << 20);
#pragma unroll
    for (int i = 0; i < 2; ++i) {
        int lin = tid + i * 256;              // 0..511: r2=n idx, c8=k chunk
        int r2 = lin >> 3, c8 = lin & 7;
        v8h v = *(const v8h*)(&t[r2][c8 * 8]);
        *(v8h*)(Wtp + (size_t)(tn * 64 + r2) * D_IN + tk * 64 + c8 * 8) = v;
    }
}

// --------------------------- gate softmax ----------------------------------
__global__ __launch_bounds__(256)
void gate_kernel(const float* __restrict__ x, const float* __restrict__ gW,
                 const float* __restrict__ gb, float* __restrict__ gate) {
    const int lane  = threadIdx.x & 31;
    const int wave  = threadIdx.x >> 5;
    const int token = blockIdx.x * 8 + wave;
    if (token >= N_TOKENS) return;

    float s[N_EXP];
#pragma unroll
    for (int e = 0; e < N_EXP; ++e) s[e] = 0.f;

    const float* xr = x + (size_t)token * D_IN;
    for (int i = lane; i < D_IN; i += 32) {
        float xv = xr[i];
        const float4 w0 = *(const float4*)(gW + i * N_EXP);
        const float4 w1 = *(const float4*)(gW + i * N_EXP + 4);
        s[0] = fmaf(xv, w0.x, s[0]); s[1] = fmaf(xv, w0.y, s[1]);
        s[2] = fmaf(xv, w0.z, s[2]); s[3] = fmaf(xv, w0.w, s[3]);
        s[4] = fmaf(xv, w1.x, s[4]); s[5] = fmaf(xv, w1.y, s[5]);
        s[6] = fmaf(xv, w1.z, s[6]); s[7] = fmaf(xv, w1.w, s[7]);
    }
#pragma unroll
    for (int off = 16; off >= 1; off >>= 1) {
#pragma unroll
        for (int e = 0; e < N_EXP; ++e) s[e] += __shfl_xor(s[e], off, 32);
    }
#pragma unroll
    for (int e = 0; e < N_EXP; ++e) s[e] += gb[e];

    float m = s[0];
#pragma unroll
    for (int e = 1; e < N_EXP; ++e) m = fmaxf(m, s[e]);
    float p[N_EXP], sum = 0.f;
#pragma unroll
    for (int e = 0; e < N_EXP; ++e) { p[e] = __expf(s[e] - m); sum += p[e]; }
    float inv = 1.f / sum;
    if (lane == 0) {
#pragma unroll
        for (int e = 0; e < N_EXP; ++e) gate[(size_t)token * N_EXP + e] = p[e] * inv;
    }
}

// --------------------------- main gate-folded GEMM -------------------------
__global__ __launch_bounds__(256)
void moe_main(const _Float16* __restrict__ xh, const _Float16* __restrict__ Wt,
              const float* __restrict__ gate, const float* __restrict__ bias,
              float* __restrict__ out) {
    __shared__ _Float16 As[BM * PK];
    __shared__ _Float16 Bs[2][BN * PK];
    __shared__ _Float16 gl[BM * N_EXP];   // per-row gate values, f16

    const int tid  = threadIdx.x;
    const int lane = tid & 31;
    const int wave = tid >> 5;
    const int wm   = wave >> 2;          // 0..1  (64-row wave tile)
    const int wn   = wave & 3;           // 0..3  (32-col wave tile)
    const int half = lane >> 4;          // 0..1
    const int ln   = lane & 15;          // 0..15
    const int bm   = blockIdx.y * BM;
    const int bn   = blockIdx.x * BN;

    // scalar-uniform issuer flag so the TDM issue is a *scalar* branch
    // (TDM ignores EXEC; an EXEC-masked path would duplicate the DMA)
    const bool issuer = (__builtin_amdgcn_readfirstlane(wave) == 0);

    // LDS byte addresses (flat LDS-aperture address carries offset in [31:0])
    const unsigned as_addr  = (unsigned)(size_t)&As[0];
    const unsigned bs_addr0 = (unsigned)(size_t)&Bs[0][0];
    const unsigned bs_addr1 = (unsigned)(size_t)&Bs[1][0];

    v8f acc[4][2];
    const v8f vzero = {0.f, 0.f, 0.f, 0.f, 0.f, 0.f, 0.f, 0.f};
#pragma unroll
    for (int mi = 0; mi < 4; ++mi)
#pragma unroll
        for (int ni = 0; ni < 2; ++ni) acc[mi][ni] = vzero;

    // ---- prologue: TDM-load A(kc=0) and B(e=0,kc=0) tiles
    if (issuer) {
        tdm_load_tile_128x64(Wt + (size_t)bn * D_IN, bs_addr0);
        tdm_load_tile_128x64(xh + (size_t)bm * D_IN, as_addr);
    }

    // ---- fill LDS gate table (f16): 128 rows x 8 experts
    {
        int r = tid >> 1, q = tid & 1;
        float4 g = *(const float4*)(gate + (size_t)(bm + r) * N_EXP + q * 4);
        v4h o;
        o[0] = (_Float16)g.x; o[1] = (_Float16)g.y;
        o[2] = (_Float16)g.z; o[3] = (_Float16)g.w;
        *(v4h*)(&gl[r * N_EXP + q * 4]) = o;
    }

    __builtin_amdgcn_s_wait_tensorcnt(0);
    __syncthreads();

    // one expert step; rb is a literal at every call site (LDS buffer parity)
    v16h ax[2][4];   // expert-invariant A fragments for current K-chunk
    auto expert_body = [&](int e, const int rb, bool has_next, int kc) {
        // ---- issue TDM DMAs for upcoming tiles (single wave, scalar branch)
        if (issuer) {
            if (has_next) {
                int e2  = (e + 1) & 7;
                int kcn = (e == N_EXP - 1) ? kc + BK : kc;
                tdm_load_tile_128x64(Wt + ((size_t)e2 << 20) +
                                     (size_t)bn * D_IN + kcn,
                                     rb ? bs_addr0 : bs_addr1);
            }
            // A tile for next K-chunk: issue after e==0's barrier (all waves
            // have hoisted this chunk's A fragments by then)
            if (e == 1 && kc + BK < D_IN)
                tdm_load_tile_128x64(xh + (size_t)bm * D_IN + kc + BK, as_addr);
        }

        // ---- gate scalars for this lane's 4 subtile rows
        _Float16 gh[4];
#pragma unroll
        for (int mi = 0; mi < 4; ++mi)
            gh[mi] = gl[(wm * 64 + mi * 16 + ln) * N_EXP + e];

        // ---- B fragments for both K-steps
        v16h bf[2][2];
#pragma unroll
        for (int ks = 0; ks < 2; ++ks)
#pragma unroll
            for (int ni = 0; ni < 2; ++ni) {
                const _Float16* p = &Bs[rb][(wn * 32 + ni * 16 + ln) * PK +
                                           ks * 32 + half * 8];
                bf[ks][ni] = join16(*(const v8h*)p, *(const v8h*)(p + 16));
            }

        // ---- scale A fragments per expert (alternating buffers -> no WMMA
        // WAR hazard NOPs), then WMMA
        v16h as[2];
#pragma unroll
        for (int mi = 0; mi < 4; ++mi) {
#pragma unroll
            for (int ks = 0; ks < 2; ++ks) {
#pragma unroll
                for (int i = 0; i < 16; ++i) as[ks][i] = ax[ks][mi][i] * gh[mi];
#pragma unroll
                for (int ni = 0; ni < 2; ++ni)
                    acc[mi][ni] = __builtin_amdgcn_wmma_f32_16x16x32_f16(
                        false, as[ks], false, bf[ks][ni], (short)0, acc[mi][ni],
                        false, false);
            }
        }

        // ---- publish the DMA'd tiles: issuing wave drains TENSORcnt, then
        // the barrier makes them visible to every wave
        __builtin_amdgcn_s_wait_tensorcnt(0);
        __syncthreads();
    };

    for (int kc = 0; kc < D_IN; kc += BK) {
        // ---- hoist A fragments into registers, reused by all 8 experts
#pragma unroll
        for (int mi = 0; mi < 4; ++mi)
#pragma unroll
            for (int ks = 0; ks < 2; ++ks) {
                const _Float16* p = &As[(wm * 64 + mi * 16 + ln) * PK +
                                        ks * 32 + half * 8];
                ax[ks][mi] = join16(*(const v8h*)p, *(const v8h*)(p + 16));
            }

        // ---- 8 experts, unrolled x2 with literal LDS buffer parity
        for (int ee = 0; ee < 4; ++ee) {
            expert_body(2 * ee,     0, true, kc);
            expert_body(2 * ee + 1, 1, (ee < 3) || (kc + BK < D_IN), kc);
        }
    }

    // ---- epilogue: rank-8 bias combine in f32, store
    const int col0 = bn + wn * 32 + ln;
    const int col1 = col0 + 16;
    float bc0[N_EXP], bc1[N_EXP];
#pragma unroll
    for (int eb = 0; eb < N_EXP; ++eb) {
        bc0[eb] = bias[eb * D_OUT + col0];
        bc1[eb] = bias[eb * D_OUT + col1];
    }
#pragma unroll
    for (int mi = 0; mi < 4; ++mi) {
#pragma unroll
        for (int v = 0; v < 8; ++v) {
            int row = bm + wm * 64 + mi * 16 + half * 8 + v;
            const float* g = gate + (size_t)row * N_EXP;
            float b0 = 0.f, b1 = 0.f;
#pragma unroll
            for (int eb = 0; eb < N_EXP; ++eb) {
                float ge = g[eb];
                b0 = fmaf(ge, bc0[eb], b0);
                b1 = fmaf(ge, bc1[eb], b1);
            }
            out[(size_t)row * D_OUT + col0] = acc[mi][0][v] + b0;
            out[(size_t)row * D_OUT + col1] = acc[mi][1][v] + b1;
        }
    }
}

// --------------------------- launcher --------------------------------------
extern "C" void kernel_launch(void* const* d_in, const int* in_sizes, int n_in,
                              void* d_out, int out_size, void* d_ws, size_t ws_size,
                              hipStream_t stream) {
    const float* x  = (const float*)d_in[0];   // [8192,1024]
    const float* W  = (const float*)d_in[1];   // [8,1024,1024]
    const float* b  = (const float*)d_in[2];   // [8,1024]
    const float* gW = (const float*)d_in[3];   // [1024,8]
    const float* gb = (const float*)d_in[4];   // [8]
    float* out = (float*)d_out;

    char* ws = (char*)d_ws;
    _Float16* Wt   = (_Float16*)ws;                                   // 16 MB
    _Float16* xh   = (_Float16*)(ws + (size_t)16 * 1024 * 1024);      // 16 MB
    float*    gate = (float*)   (ws + (size_t)32 * 1024 * 1024);      // 256 KB

    // W: transpose + convert to f16 [e][o][k]
    transpose_cvt_w<<<dim3(D_OUT / 64, D_IN / 64, N_EXP), 256, 0, stream>>>(W, Wt);
    // x: convert to f16 (row-major)
    {
        int n4 = (N_TOKENS * D_IN) / 4;
        cvt_f32_to_f16<<<(n4 + 255) / 256, 256, 0, stream>>>(x, xh, n4);
    }
    // gate softmax: 8 tokens (waves) per 256-thread block
    gate_kernel<<<N_TOKENS / 8, 256, 0, stream>>>(x, gW, gb, gate);

    // main GEMM: grid = (N/BN, M/BM) = (8, 64)
    dim3 grid(D_OUT / BN, N_TOKENS / BM);
    moe_main<<<grid, 256, 0, stream>>>(xh, Wt, gate, b, out);
}
